// SoftAlignAttention_2147483648218
// MI455X (gfx1250) — matrix operand order repaired and checked by reading, not compiled, not run on hardware
//
#include <hip/hip_runtime.h>
#include <hip/hip_bf16.h>
#include <math.h>

typedef __attribute__((ext_vector_type(16))) __bf16 v16bf;
typedef __attribute__((ext_vector_type(8)))  __bf16 v8bf;
typedef __attribute__((ext_vector_type(2)))  __bf16 v2bf;
typedef __attribute__((ext_vector_type(8)))  float  v8f;

#define NB   32      // batch
#define NS   512     // N == M
#define DK   1024    // D
#define BM   128     // block tile rows (4 wave-row groups of 32)
#define BN   64      // block tile cols (2 wave-col groups of 32)
#define BK   32      // K step (matches wmma 16x16x32 bf16)

// Build the 16x32 bf16 A-fragment for one lane from a row-major LDS row.
// Lane l (l<16): row=l, K = {0..7, 16..23}; lane l>=16: row=l-16, K = {8..15, 24..31}.
__device__ __forceinline__ v16bf make_afrag(const __bf16* rowp, int kbase) {
  v8bf lo = *(const v8bf*)(rowp + kbase);        // K = kbase .. kbase+7
  v8bf hi = *(const v8bf*)(rowp + kbase + 16);   // K = kbase+16 .. kbase+23
  return __builtin_shufflevector(lo, hi, 0, 1, 2, 3, 4, 5, 6, 7,
                                         8, 9, 10, 11, 12, 13, 14, 15);
}

__device__ __forceinline__ void cvt_store8(__bf16* dst, float4 f0, float4 f1) {
  v2bf* d = (v2bf*)dst;
  d[0] = (v2bf){(__bf16)f0.x, (__bf16)f0.y};
  d[1] = (v2bf){(__bf16)f0.z, (__bf16)f0.w};
  d[2] = (v2bf){(__bf16)f1.x, (__bf16)f1.y};
  d[3] = (v2bf){(__bf16)f1.z, (__bf16)f1.w};
}

// ---------------------------------------------------------------------------
// Kernel 1: S[b,n,m] = dot(x1[b,n,:], x2[b,m,:])  (bf16 WMMA, fp32 accum)
// Block tile 128x64, 8 waves, each wave owns a 32x32 output (2x2 WMMA tiles).
// ---------------------------------------------------------------------------
__global__ __launch_bounds__(256) void scores_kernel(const float* __restrict__ x1,
                                                     const float* __restrict__ x2,
                                                     float* __restrict__ S) {
  const int b       = blockIdx.z;
  const int rowBase = blockIdx.y * BM;   // n
  const int colBase = blockIdx.x * BN;   // m
  __shared__ __align__(64) __bf16 As[BM][BK];
  __shared__ __align__(64) __bf16 Bs[BN][BK];

  const int tid  = threadIdx.x;          // 256 threads = 8 waves
  const int lane = tid & 31;
  const int w    = tid >> 5;
  const int wr   = (w & 3) * 32;         // wave row base: 0,32,64,96
  const int wc   = (w >> 2) * 32;        // wave col base: 0,32
  const int l16  = lane & 15;
  const int half = lane >> 4;

  const float* X1 = x1 + (size_t)b * NS * DK;
  const float* X2 = x2 + (size_t)b * NS * DK;

  v8f acc00 = {}, acc01 = {}, acc10 = {}, acc11 = {};

  for (int k0 = 0; k0 < DK; k0 += BK) {
    { // stage A tile 128x32 (x1 rows): thread -> 16 elements
      int r = tid >> 1; int kq = (tid & 1) * 16;
      const float4* p = (const float4*)(X1 + (size_t)(rowBase + r) * DK + k0 + kq);
      float4 f0 = p[0], f1 = p[1], f2 = p[2], f3 = p[3];
      cvt_store8(&As[r][kq], f0, f1);
      cvt_store8(&As[r][kq + 8], f2, f3);
    }
    { // stage B tile 64x32 (x2 rows = columns of x2^T): thread -> 8 elements
      int r = tid >> 2; int kq = (tid & 3) * 8;
      const float4* p = (const float4*)(X2 + (size_t)(colBase + r) * DK + k0 + kq);
      cvt_store8(&Bs[r][kq], p[0], p[1]);
    }
    __syncthreads();
    v16bf a0 = make_afrag(&As[wr + l16][0], half * 8);
    v16bf a1 = make_afrag(&As[wr + 16 + l16][0], half * 8);
    v16bf b0 = make_afrag(&Bs[wc + l16][0], half * 8);
    v16bf b1 = make_afrag(&Bs[wc + 16 + l16][0], half * 8);
    acc00 = __builtin_amdgcn_wmma_f32_16x16x32_bf16(false, a0, false, b0, (short)0, acc00, false, false);
    acc01 = __builtin_amdgcn_wmma_f32_16x16x32_bf16(false, a0, false, b1, (short)0, acc01, false, false);
    acc10 = __builtin_amdgcn_wmma_f32_16x16x32_bf16(false, a1, false, b0, (short)0, acc10, false, false);
    acc11 = __builtin_amdgcn_wmma_f32_16x16x32_bf16(false, a1, false, b1, (short)0, acc11, false, false);
    __syncthreads();
  }

  // C layout: VGPR v, lanes 0-15 -> M=v, lanes 16-31 -> M=v+8, N=lane%16
  float* o0 = S + ((size_t)b * NS + rowBase + wr + half * 8) * NS + colBase + wc + l16;
  float* o1 = S + ((size_t)b * NS + rowBase + wr + 16 + half * 8) * NS + colBase + wc + l16;
#pragma unroll
  for (int v = 0; v < 8; ++v) {
    o0[(size_t)v * NS]      = acc00[v];
    o0[(size_t)v * NS + 16] = acc01[v];
    o1[(size_t)v * NS]      = acc10[v];
    o1[(size_t)v * NS + 16] = acc11[v];
  }
}

// ---------------------------------------------------------------------------
// Kernel 2: per-row masked max & sum-exp (one wave32 per row, online softmax)
// ---------------------------------------------------------------------------
__global__ __launch_bounds__(256) void row_stats_kernel(const float* __restrict__ S,
                                                        const int* __restrict__ mask,
                                                        float* __restrict__ rmax,
                                                        float* __restrict__ rsum) {
  const int lane = threadIdx.x & 31;
  const int w    = threadIdx.x >> 5;
  const int row  = blockIdx.x * 8 + w;           // 0 .. NB*NS-1
  const int b    = row / NS;
  const float* p  = S + (size_t)row * NS;
  const int*   mk = mask + (size_t)b * NS;

  float m = -__builtin_inff(), s = 0.f;
  for (int j = lane; j < NS; j += 32) {
    if (mk[j]) {
      float v = p[j];
      if (v > m) { s = s * __expf(m - v) + 1.f; m = v; }
      else       { s += __expf(v - m); }
    }
  }
#pragma unroll
  for (int off = 16; off >= 1; off >>= 1) {
    float mo = __shfl_xor(m, off, 32);
    float so = __shfl_xor(s, off, 32);
    float mm = fmaxf(m, mo);
    float s1 = (m  == -__builtin_inff()) ? 0.f : s  * __expf(m  - mm);
    float s2 = (mo == -__builtin_inff()) ? 0.f : so * __expf(mo - mm);
    m = mm; s = s1 + s2;
  }
  if (lane == 0) { rmax[row] = m; rsum[row] = s; }
}

// ---------------------------------------------------------------------------
// Kernel 3: per-column masked max & sum-exp (thread per column, coalesced)
// ---------------------------------------------------------------------------
__global__ __launch_bounds__(256) void col_stats_kernel(const float* __restrict__ S,
                                                        const int* __restrict__ mask,
                                                        float* __restrict__ cmax,
                                                        float* __restrict__ csum) {
  const int gid = blockIdx.x * blockDim.x + threadIdx.x;  // (b,m) flat
  const int b = gid / NS;
  const int m = gid % NS;
  const float* p  = S + (size_t)b * NS * NS + m;
  const int*   mk = mask + (size_t)b * NS;

  float mx = -__builtin_inff(), s = 0.f;
  for (int n = 0; n < NS; ++n) {
    if (mk[n]) {
      float v = p[(size_t)n * NS];
      if (v > mx) { s = s * __expf(mx - v) + 1.f; mx = v; }
      else        { s += __expf(v - mx); }
    }
  }
  cmax[gid] = mx; csum[gid] = s;
}

// ---------------------------------------------------------------------------
// Kernel 4/5: out[b,row,d] = sum_k softmax(S)[row,k] * X[k,d]
// transA=0: A[row][k] = S[row][k]   (dir 1, rows=n, k=m, X=x2, mask=mask2)
// transA=1: A[row][k] = S[k][row]   (dir 2, rows=m, k=n, X=x1, mask=mask1)
// ---------------------------------------------------------------------------
__global__ __launch_bounds__(256) void pv_kernel(const float* __restrict__ S,
                                                 const float* __restrict__ X,
                                                 const int* __restrict__ maskK,
                                                 const float* __restrict__ rmax,
                                                 const float* __restrict__ rsum,
                                                 float* __restrict__ out,
                                                 int transA) {
  const int b       = blockIdx.z;
  const int rowBase = blockIdx.y * BM;   // output row
  const int colBase = blockIdx.x * BN;   // d
  __shared__ __align__(64) __bf16 As[BM][BK];
  __shared__ __align__(64) __bf16 Bs[BN][BK];
  __shared__ float s_max[BM], s_inv[BM];

  const int tid  = threadIdx.x;
  const int lane = tid & 31;
  const int w    = tid >> 5;
  const int wr   = (w & 3) * 32;
  const int wc   = (w >> 2) * 32;
  const int l16  = lane & 15;
  const int half = lane >> 4;

  if (tid < BM) {
    float m = rmax[(size_t)b * NS + rowBase + tid];
    float s = rsum[(size_t)b * NS + rowBase + tid];
    s_max[tid] = m;
    s_inv[tid] = (s > 0.f) ? (1.f / s) : 0.f;
  }
  __syncthreads();

  const float* Sb = S + (size_t)b * NS * NS;
  const float* Xb = X + (size_t)b * NS * DK;
  const int*   mk = maskK + (size_t)b * NS;

  v8f acc00 = {}, acc01 = {}, acc10 = {}, acc11 = {};

  for (int k0 = 0; k0 < NS; k0 += BK) {
    if (!transA) {
      // stage A 128x32: read S[row][k] row-major, softmax on the fly
      int r = tid >> 1; int kq = (tid & 1) * 16;
      const float4* p = (const float4*)(Sb + (size_t)(rowBase + r) * NS + k0 + kq);
      float4 f0 = p[0], f1 = p[1], f2 = p[2], f3 = p[3];
      float mv = s_max[r], iv = s_inv[r];
      float vv[16] = {f0.x, f0.y, f0.z, f0.w, f1.x, f1.y, f1.z, f1.w,
                      f2.x, f2.y, f2.z, f2.w, f3.x, f3.y, f3.z, f3.w};
#pragma unroll
      for (int j = 0; j < 16; ++j) {
        int km = k0 + kq + j;
        As[r][kq + j] = (__bf16)(mk[km] ? __expf(vv[j] - mv) * iv : 0.f);
      }
    } else {
      // transposed: A[row][k] = S[k][row]; contiguous along row -> coalesced
      int k = tid >> 3; int rq = (tid & 7) * 16;
      const float4* p = (const float4*)(Sb + (size_t)(k0 + k) * NS + rowBase + rq);
      float4 f0 = p[0], f1 = p[1], f2 = p[2], f3 = p[3];
      int mval = mk[k0 + k];
      float vv[16] = {f0.x, f0.y, f0.z, f0.w, f1.x, f1.y, f1.z, f1.w,
                      f2.x, f2.y, f2.z, f2.w, f3.x, f3.y, f3.z, f3.w};
#pragma unroll
      for (int j = 0; j < 16; ++j) {
        int r = rq + j;
        As[r][k] = (__bf16)(mval ? __expf(vv[j] - s_max[r]) * s_inv[r] : 0.f);
      }
    }
    { // stage B transposed: Bs[n][k] = X[k0+k][colBase+n]; global coalesced on n
      int k = tid >> 3; int nq = (tid & 7) * 8;
      const float4* p = (const float4*)(Xb + (size_t)(k0 + k) * DK + colBase + nq);
      float4 f0 = p[0], f1 = p[1];
      float vv[8] = {f0.x, f0.y, f0.z, f0.w, f1.x, f1.y, f1.z, f1.w};
#pragma unroll
      for (int j = 0; j < 8; ++j) Bs[nq + j][k] = (__bf16)vv[j];
    }
    __syncthreads();
    v16bf a0 = make_afrag(&As[wr + l16][0], half * 8);
    v16bf a1 = make_afrag(&As[wr + 16 + l16][0], half * 8);
    v16bf b0 = make_afrag(&Bs[wc + l16][0], half * 8);
    v16bf b1 = make_afrag(&Bs[wc + 16 + l16][0], half * 8);
    acc00 = __builtin_amdgcn_wmma_f32_16x16x32_bf16(false, a0, false, b0, (short)0, acc00, false, false);
    acc01 = __builtin_amdgcn_wmma_f32_16x16x32_bf16(false, a0, false, b1, (short)0, acc01, false, false);
    acc10 = __builtin_amdgcn_wmma_f32_16x16x32_bf16(false, a1, false, b0, (short)0, acc10, false, false);
    acc11 = __builtin_amdgcn_wmma_f32_16x16x32_bf16(false, a1, false, b1, (short)0, acc11, false, false);
    __syncthreads();
  }

  float* o0 = out + ((size_t)b * NS + rowBase + wr + half * 8) * DK + colBase + wc + l16;
  float* o1 = out + ((size_t)b * NS + rowBase + wr + 16 + half * 8) * DK + colBase + wc + l16;
#pragma unroll
  for (int v = 0; v < 8; ++v) {
    o0[(size_t)v * DK]      = acc00[v];
    o0[(size_t)v * DK + 16] = acc01[v];
    o1[(size_t)v * DK]      = acc10[v];
    o1[(size_t)v * DK + 16] = acc11[v];
  }
}

// ---------------------------------------------------------------------------
extern "C" void kernel_launch(void* const* d_in, const int* in_sizes, int n_in,
                              void* d_out, int out_size, void* d_ws, size_t ws_size,
                              hipStream_t stream) {
  (void)in_sizes; (void)n_in; (void)out_size; (void)ws_size;
  const float* x1    = (const float*)d_in[0];
  const float* x2    = (const float*)d_in[1];
  const int*   mask1 = (const int*)d_in[2];
  const int*   mask2 = (const int*)d_in[3];

  float* S = (float*)d_ws;                       // 32 * 512 * 512 fp32 = 32 MB
  size_t sElems = (size_t)NB * NS * NS;
  float* rmax1 = S + sElems;                     // per-n stats (dir 1)
  float* rsum1 = rmax1 + (size_t)NB * NS;
  float* rmax2 = rsum1 + (size_t)NB * NS;        // per-m stats (dir 2)
  float* rsum2 = rmax2 + (size_t)NB * NS;

  float* out1 = (float*)d_out;                   // [B, N, D]
  float* out2 = out1 + (size_t)NB * NS * DK;     // [B, M, D]

  dim3 blk(256);
  scores_kernel<<<dim3(NS / BN, NS / BM, NB), blk, 0, stream>>>(x1, x2, S);
  row_stats_kernel<<<dim3(NB * NS / 8), blk, 0, stream>>>(S, mask2, rmax1, rsum1);
  col_stats_kernel<<<dim3(NB * NS / 256), blk, 0, stream>>>(S, mask1, rmax2, rsum2);
  pv_kernel<<<dim3(DK / BN, NS / BM, NB), blk, 0, stream>>>(S, x2, mask2, rmax1, rsum1, out1, 0);
  pv_kernel<<<dim3(DK / BN, NS / BM, NB), blk, 0, stream>>>(S, x1, mask1, rmax2, rsum2, out2, 1);
}
